// MyNet_87093346828557
// MI455X (gfx1250) — matrix operand order repaired and checked by reading, not compiled, not run on hardware
//
#include <hip/hip_runtime.h>

#if __has_builtin(__builtin_amdgcn_tensor_load_to_lds)
  #define HAVE_TDM 1
  #if __has_include(<hip/amd_detail/amd_gfx1250_TDM.h>)
    #define TDM_6ARG 1
  #endif
#endif

typedef __attribute__((ext_vector_type(16))) __bf16        v16bf;
typedef __attribute__((ext_vector_type(8)))  float         v8f;
typedef __attribute__((ext_vector_type(4)))  unsigned int  v4u;
typedef __attribute__((ext_vector_type(8)))  int           v8i;
typedef __attribute__((ext_vector_type(4)))  int           v4i;

union Frag { v16bf v; uint4 q[2]; };

constexpr int HW      = 512;
constexpr int CIN     = 8;
constexpr int WSTRIDE = 104;        // 12 pos * 8 ci + 8 pad -> bank-conflict-free B loads
constexpr int RSTR    = 130 * 8;    // s_in row stride in ushorts (130 x positions, ci-minor)

__device__ __forceinline__ unsigned short f32_to_bf16_rne(float f) {
    union { float f; unsigned u; } cv; cv.f = f;
    unsigned u = cv.u;
    unsigned r = u + 0x7FFFu + ((u >> 16) & 1u);
    return (unsigned short)(r >> 16);
}

// ---- weight prep: f32 [co][ci][3][3] -> bf16 [co][pos(12, 9..11 zero)][ci] (+pad to WSTRIDE)
// This is the exact LDS image the conv kernel wants, so the conv kernel can TDM it in 1:1.
template<int COUT>
__global__ __launch_bounds__(256) void prep_weights(const float* __restrict__ w,
                                                    unsigned short* __restrict__ out) {
    constexpr int COPAD = (COUT < 16) ? 16 : COUT;
    for (int t = threadIdx.x; t < COPAD * WSTRIDE; t += 256) {
        const int co  = t / WSTRIDE;
        const int rem = t - co * WSTRIDE;
        const int pos = rem >> 3;
        const int ci  = rem & 7;
        unsigned short v = 0;
        if (co < COUT && pos < 9)
            v = f32_to_bf16_rne(w[(co * CIN + ci) * 9 + pos]);
        out[t] = v;
    }
}

// 3x3 conv, stride 1, pad 1, Cin=8, H=W=512, implicit GEMM on bf16 WMMA.
// Block = 128 consecutive x of one (n,y) row; M-tile 16 px, N-tile 16 ch, K = pos*8+ci (96).
// IN_F32_NCHW:  input is f32 NCHW (conv1), else bf16 NHWC (ci-minor).
// OUT_F32_NCHW: output is f32 NCHW (conv3), else bf16 NHWC via LDS-staged coalesced store.
template<int COUT, bool IN_F32_NCHW, bool OUT_F32_NCHW>
__global__ __launch_bounds__(256) void conv3x3_wmma(const void* __restrict__ inp,
                                                    const unsigned short* __restrict__ wprep,
                                                    void* __restrict__ outp)
{
    constexpr int COPAD = (COUT < 16) ? 16 : COUT;
    constexpr int NT    = COPAD / 16;

    __shared__ unsigned short s_in[3 * RSTR];                       // [row(3)][x(130)][ci(8)]
    __shared__ unsigned short s_w[COPAD * WSTRIDE];                 // [co][pos(12)][ci]
    __shared__ unsigned short s_out[OUT_F32_NCHW ? 8 : 128 * 8];    // NHWC out tile

    const int tid  = threadIdx.x;
    const int x0   = blockIdx.x * 128;
    const int y    = blockIdx.y;
    const int n    = blockIdx.z;
    const int lane = tid & 31;
    const int wave = tid >> 5;
    const uint4 zq = make_uint4(0u, 0u, 0u, 0u);

    // ---- weights into LDS: TDM DMA of the pre-built image (fallback: vector copy)
#if HAVE_TDM
    if (wave == 0) {
        const unsigned lds_base     = (unsigned)(size_t)(const void*)&s_w[0];
        const unsigned long long ga = (unsigned long long)(size_t)(const void*)wprep;
        constexpr unsigned N64 = (unsigned)(COPAD * WSTRIDE * 2 / 8);  // 8B elements
        v4u g0;
        g0[0] = 1u;                                                // count=1, user mode
        g0[1] = lds_base;                                          // lds_addr
        g0[2] = (unsigned)ga;                                      // global_addr[31:0]
        g0[3] = ((unsigned)(ga >> 32) & 0x1FFFFFFu) | (2u << 30);  // addr[56:32] | type=2
        v8i g1;
        g1[0] = (int)(3u << 16);                  // workgroup_mask=0, data_size=8B
        g1[1] = (int)((N64 & 0xFFFFu) << 16);     // tensor_dim0[15:0]
        g1[2] = (int)((N64 >> 16) | (1u << 16));  // tensor_dim0[31:16] | tensor_dim1=1
        g1[3] = (int)(N64 << 16);                 // tile_dim0 = N64 (1-D tile)
        g1[4] = 0;                                // tile_dim1/2 = 0 (unused)
        g1[5] = (int)N64;                         // tensor_dim0_stride
        g1[6] = 0;
        g1[7] = 0;
        const v4i z4 = {0, 0, 0, 0};
#ifdef TDM_6ARG
        const v8i z8 = {0, 0, 0, 0, 0, 0, 0, 0};
        __builtin_amdgcn_tensor_load_to_lds(g0, g1, z4, z4, z8, 0);
#else
        __builtin_amdgcn_tensor_load_to_lds(g0, g1, z4, z4, 0);
#endif
    }
#else
    for (int t = tid; t < COPAD * WSTRIDE / 8; t += 256)
        ((uint4*)s_w)[t] = ((const uint4*)wprep)[t];
#endif

    // ---- stage input rows y-1..y+1, x0-1..x0+128 (zero pad at borders), ci-minor in LDS
    if (IN_F32_NCHW) {
        const float* g = (const float*)inp;
        for (int t = tid; t < 3 * 8 * 130; t += 256) {
            const int xl  = t % 130;
            const int ci  = (t / 130) & 7;
            const int row = t / (130 * 8);
            const int gx  = x0 - 1 + xl;
            const int gy  = y - 1 + row;
            unsigned short v = 0;
            if (gx >= 0 && gx < HW && gy >= 0 && gy < HW)
                v = f32_to_bf16_rne(g[(((size_t)n * CIN + ci) * HW + gy) * HW + gx]);
            s_in[row * RSTR + xl * 8 + ci] = v;
        }
    } else {
        // NHWC bf16: one pixel = 8 ci = 16B chunk -> pure b128 global->LDS pipe
        const unsigned short* g = (const unsigned short*)inp;
        for (int c = tid; c < 3 * 130; c += 256) {
            const int row = c / 130;
            const int xl  = c - row * 130;
            const int gx  = x0 - 1 + xl;
            const int gy  = y - 1 + row;
            uint4 v = zq;
            if (gx >= 0 && gx < HW && gy >= 0 && gy < HW)
                v = *(const uint4*)&g[(((size_t)n * HW + gy) * HW + gx) * 8];
            *(uint4*)&s_in[row * RSTR + xl * 8] = v;
        }
    }
#if HAVE_TDM
    if (wave == 0) __builtin_amdgcn_s_wait_tensorcnt(0);
#endif
    __syncthreads();

    // ---- compute: wave w owns M-tile w; loops over N-tiles
    const int half = lane >> 4;
    const int l    = lane & 15;
    const int mt   = wave;
    const int pxl  = mt * 16 + l;   // A-row pixel (local x)

    // A fragments (ISA 16-bit A 16x32): lane<16 holds K 0-7 & 16-23, lane>=16 K 8-15 & 24-31
    Frag a[3];
#pragma unroll
    for (int kk = 0; kk < 3; ++kk) {
        const int p0 = 4 * kk + half;
        const int p1 = p0 + 2;
        a[kk].q[0] = (p0 <= 8)
            ? *(const uint4*)&s_in[(p0 / 3) * RSTR + (pxl + p0 % 3) * 8] : zq;
        a[kk].q[1] = (p1 <= 8)
            ? *(const uint4*)&s_in[(p1 / 3) * RSTR + (pxl + p1 % 3) * 8] : zq;
    }

#pragma unroll
    for (int nt = 0; nt < NT; ++nt) {
        const int co = nt * 16 + l;
        v8f acc = {};
#pragma unroll
        for (int kk = 0; kk < 3; ++kk) {
            const int pb0 = 4 * kk + half;  // B: lane<16 K 0-7 & 16-23, lane>=16 K 8-15 & 24-31
            Frag b;
            b.q[0] = *(const uint4*)&s_w[co * WSTRIDE + (pb0 << 3)];
            b.q[1] = *(const uint4*)&s_w[co * WSTRIDE + ((pb0 + 2) << 3)];
            acc = __builtin_amdgcn_wmma_f32_16x16x32_bf16(
                false, a[kk].v, false, b.v, (short)0, acc, false, false);
        }

        if (OUT_F32_NCHW) {
            // lane holds column co; vgpr r -> pixel x0 + mt*16 + 8*half + r (coalesced f32x4)
            float* o = (float*)outp;
            const int px = x0 + mt * 16 + half * 8;
            const size_t base = (((size_t)n * COUT + co) * HW + y) * HW + px;
            *(float4*)(o + base)     = make_float4(acc[0], acc[1], acc[2], acc[3]);
            *(float4*)(o + base + 4) = make_float4(acc[4], acc[5], acc[6], acc[7]);
        } else {
            if (l < CIN) {   // real channels only (COUT==8 padded to 16)
#pragma unroll
                for (int r = 0; r < 8; ++r)
                    s_out[(mt * 16 + half * 8 + r) * 8 + l] = f32_to_bf16_rne(acc[r]);
            }
        }
    }

    if (!OUT_F32_NCHW) {
        __syncthreads();
        if (tid < 128) {   // 2KB contiguous NHWC tile: one b128 store per thread
            unsigned short* o = (unsigned short*)outp;
            const size_t base = (((size_t)n * HW + y) * HW + x0 + tid) * 8;
            *(uint4*)(o + base) = *(const uint4*)&s_out[tid * 8];
        }
    }
}

extern "C" void kernel_launch(void* const* d_in, const int* in_sizes, int n_in,
                              void* d_out, int out_size, void* d_ws, size_t ws_size,
                              hipStream_t stream) {
    (void)in_sizes; (void)n_in; (void)out_size; (void)ws_size;
    const float* x  = (const float*)d_in[0];
    const float* w1 = (const float*)d_in[1];
    const float* w2 = (const float*)d_in[2];
    const float* w3 = (const float*)d_in[3];

    // h1 (bf16 NHWC, 67 MB) lives inside d_out (dead before conv3 overwrites it);
    // h2 (bf16 NHWC, 67 MB) + prepped weights live in d_ws. Both fit the 192 MB L2.
    unsigned short* h1 = (unsigned short*)d_out;
    unsigned short* h2 = (unsigned short*)d_ws;
    char* wsb = (char*)d_ws;
    const size_t H2_BYTES = (size_t)16 * HW * HW * CIN * 2;
    unsigned short* w1p = (unsigned short*)(wsb + H2_BYTES);
    unsigned short* w2p = (unsigned short*)(wsb + H2_BYTES + 8192);
    unsigned short* w3p = (unsigned short*)(wsb + H2_BYTES + 16384);

    prep_weights<8 ><<<dim3(1), dim3(256), 0, stream>>>(w1, w1p);
    prep_weights<8 ><<<dim3(1), dim3(256), 0, stream>>>(w2, w2p);
    prep_weights<64><<<dim3(1), dim3(256), 0, stream>>>(w3, w3p);

    dim3 grid(HW / 128, HW, 16);   // x-blocks, y, n
    dim3 block(256);
    conv3x3_wmma<8,  true,  false><<<grid, block, 0, stream>>>(x,  w1p, h1);
    conv3x3_wmma<8,  false, false><<<grid, block, 0, stream>>>(h1, w2p, h2);
    conv3x3_wmma<64, false, true ><<<grid, block, 0, stream>>>(h2, w3p, d_out);
}
